// MultiHeadSeqAttention_59579786330679
// MI455X (gfx1250) — matrix operand order repaired and checked by reading, not compiled, not run on hardware
//
#include <hip/hip_runtime.h>
#include <math.h>

typedef unsigned short u16;
typedef unsigned int   u32;
typedef __attribute__((ext_vector_type(16))) __bf16 v16bf;
typedef __attribute__((ext_vector_type(8)))  float  v8f;
typedef __attribute__((ext_vector_type(8)))  u32    v8u;
typedef __attribute__((address_space(3)))    u16    lds_u16;

#define HID   1024
#define HEADS 16
#define HDIM  64
#define BATCH 2
#define SEQ   2048
#define MROWS (BATCH * SEQ)              // 4096
#define NQ_ELEMS ((size_t)MROWS * HID)   // 4 Mi
#define NW_ELEMS ((size_t)HID * HID)     // 1 Mi

// ---------------- helpers ----------------

__device__ __forceinline__ u16 f2bf(float f) {
    u32 u = __float_as_uint(f);
    u = (u + 0x7FFFu + ((u >> 16) & 1u)) >> 16;   // RNE
    return (u16)u;
}

// Load a 16-element bf16 fragment as two 16-byte chunks (global or LDS).
__device__ __forceinline__ v16bf ldfrag(const u16* p, int o0, int o1) {
    uint4 a = *(const uint4*)(p + o0);
    uint4 b = *(const uint4*)(p + o1);
    v8u r;
    r[0] = a.x; r[1] = a.y; r[2] = a.z; r[3] = a.w;
    r[4] = b.x; r[5] = b.y; r[6] = b.z; r[7] = b.w;
    return __builtin_bit_cast(v16bf, r);
}

__device__ __forceinline__ v8f wmma_bf16(v16bf a, v16bf b, v8f c) {
    return __builtin_amdgcn_wmma_f32_16x16x32_bf16(
        false, a, false, b, (short)0, c, false, false);
}

// Async global -> LDS 16-byte copy (per lane), tracked by ASYNCcnt.
__device__ __forceinline__ void async_b128(u16* lds_dst, const u16* gsrc) {
    lds_u16* l = (lds_u16*)lds_dst;
    asm volatile("global_load_async_to_lds_b128 %0, %1, off"
                 :: "v"(l), "v"(gsrc) : "memory");
}

// ---------------- fp32 -> bf16 conversion ----------------

__global__ void cvt_bf16_kernel(const float* __restrict__ src,
                                u16* __restrict__ dst, int n) {
    int i = blockIdx.x * blockDim.x + threadIdx.x;
    if (i < n) dst[i] = f2bf(src[i]);
}

// ---------------- generic GEMM: C[M,N] = A[M,K] * W[N,K]^T ----------------
// A, W bf16; accumulate f32.
// MODE 0: bf16 row-major [M,N]
// MODE 1: bf16 "V-transposed": out[((b*16+h)*64+d)*2048 + m]
// MODE 2: fp32 row-major [M,N]
// MODE 3: bf16 row-major [M,N], scaled by 0.125 (softmax scale folded into Q)
// Block: 256 threads = 8 waves (4 M x 2 N), block tile 128x128, wave tile 32x64.

template <int MODE>
__global__ void __launch_bounds__(256, 1)
gemm_bf16_kernel(const u16* __restrict__ A,
                 const u16* __restrict__ W,
                 void* __restrict__ out) {
    const int K = HID, N = HID;
    int lane = threadIdx.x & 31;
    int w    = threadIdx.x >> 5;
    int wm = w & 3, wn = w >> 2;
    int rbase = blockIdx.y * 128 + wm * 32;
    int cbase = blockIdx.x * 128 + wn * 64;
    int l16 = lane & 15;
    int lh  = (lane >> 4) & 1;

    v8f c00 = {}, c01 = {}, c02 = {}, c03 = {};
    v8f c10 = {}, c11 = {}, c12 = {}, c13 = {};

    const int aoff0 = (rbase + l16) * K + lh * 16;
    const int aoff1 = (rbase + 16 + l16) * K + lh * 16;
    const int boff0 = (cbase +  0 + l16) * K + lh * 16;
    const int boff1 = (cbase + 16 + l16) * K + lh * 16;
    const int boff2 = (cbase + 32 + l16) * K + lh * 16;
    const int boff3 = (cbase + 48 + l16) * K + lh * 16;

    for (int kk = 0; kk < K / 32; ++kk) {
        int kb = kk * 32;
        if (kk + 1 < K / 32) {
            __builtin_prefetch(A + aoff0 + kb + 32, 0, 1);
            __builtin_prefetch(W + boff0 + kb + 32, 0, 1);
        }
        v16bf a0 = ldfrag(A, aoff0 + kb, aoff0 + kb + 8);
        v16bf a1 = ldfrag(A, aoff1 + kb, aoff1 + kb + 8);
        v16bf b0 = ldfrag(W, boff0 + kb, boff0 + kb + 8);
        v16bf b1 = ldfrag(W, boff1 + kb, boff1 + kb + 8);
        v16bf b2 = ldfrag(W, boff2 + kb, boff2 + kb + 8);
        v16bf b3 = ldfrag(W, boff3 + kb, boff3 + kb + 8);
        c00 = wmma_bf16(a0, b0, c00);
        c10 = wmma_bf16(a1, b0, c10);
        c01 = wmma_bf16(a0, b1, c01);
        c11 = wmma_bf16(a1, b1, c11);
        c02 = wmma_bf16(a0, b2, c02);
        c12 = wmma_bf16(a1, b2, c12);
        c03 = wmma_bf16(a0, b3, c03);
        c13 = wmma_bf16(a1, b3, c13);
    }

    // C layout: VGPR v -> row (v + 8*lh) within 16-row tile, lane%16 -> col.
    #pragma unroll
    for (int tm = 0; tm < 2; ++tm) {
        #pragma unroll
        for (int tn = 0; tn < 4; ++tn) {
            v8f acc = tm == 0
                ? (tn == 0 ? c00 : tn == 1 ? c01 : tn == 2 ? c02 : c03)
                : (tn == 0 ? c10 : tn == 1 ? c11 : tn == 2 ? c12 : c13);
            #pragma unroll
            for (int v = 0; v < 8; ++v) {
                int m = rbase + tm * 16 + v + lh * 8;
                int n = cbase + tn * 16 + l16;
                float val = acc[v];
                if (MODE == 0) {
                    ((u16*)out)[m * N + n] = f2bf(val);
                } else if (MODE == 1) {
                    int h = n >> 6, d = n & 63;
                    int b = m >> 11, mm = m & 2047;
                    ((u16*)out)[(((b * HEADS + h) * HDIM + d) << 11) + mm] = f2bf(val);
                } else if (MODE == 2) {
                    ((float*)out)[m * N + n] = val;
                } else {
                    ((u16*)out)[m * N + n] = f2bf(val * 0.125f);
                }
            }
        }
    }
}

// ---------------- flash attention ----------------
// Qp (pre-scaled by 1/sqrt(64)), Kp: bf16 [B, SEQ, HID]. Vt: bf16 [B, H, 64, SEQ].
// Out: bf16 [B, SEQ, HID]. Block = 256 threads = 8 waves, each wave owns 16 query
// rows. K/V chunks of 64 columns staged in a 4-deep LDS ring via async loads
// (amortizes softmax reductions/barriers over 2x columns vs 32-wide chunks).

#define CW  64           // chunk width (columns)
#define NCH (SEQ / CW)   // 32 chunks

__global__ void __launch_bounds__(256, 1)
attn_kernel(const u16* __restrict__ Qp,
            const u16* __restrict__ Kp,
            const u16* __restrict__ Vt,
            u16* __restrict__ Out) {
    __shared__ u16 kbuf[4][CW * 64];   // K chunk: [n_local][d]      (8 KB each)
    __shared__ u16 vbuf[4][64 * CW];   // V chunk: [d][n_local]      (8 KB each)
    __shared__ u16 plds[8][16 * CW];   // per-wave P transpose tile  (2 KB each)

    int tid  = threadIdx.x;
    int lane = tid & 31;
    int w    = tid >> 5;
    int l16 = lane & 15;
    int lh  = (lane >> 4) & 1;
    int mbase = blockIdx.x * 128 + w * 16;
    int h = blockIdx.y;
    int b = blockIdx.z;

    const u16* kpp = Kp + (size_t)b * SEQ * HID + h * HDIM;
    const u16* vtp = Vt + ((size_t)(b * HEADS + h) * HDIM) * SEQ;
    u16* pw = plds[w];

    // Per-thread async staging: 64 rows x 64 cols = 8 KB = 2 x b128 per thread.
    int sr = tid >> 2, sc = (tid & 3) * 16;   // row 0..63, col base 0..48

    // Q fragments (K = 64 -> 2 ksteps of 32), held for the whole pass.
    const int qrow = (b * SEQ + mbase + l16) * HID + h * HDIM;
    v16bf q0 = ldfrag(Qp, qrow + lh * 16,      qrow + lh * 16 + 8);
    v16bf q1 = ldfrag(Qp, qrow + 32 + lh * 16, qrow + 32 + lh * 16 + 8);

    float mrow[8], lrow[8];
    #pragma unroll
    for (int v = 0; v < 8; ++v) { mrow[v] = -INFINITY; lrow[v] = 0.f; }
    v8f o0 = {}, o1 = {}, o2 = {}, o3 = {};

    // Prologue: issue chunks 0 and 1 (4 async ops per chunk per wave).
    #pragma unroll
    for (int p = 0; p < 2; ++p) {
        int nbase = p * CW;
        async_b128(kbuf[p] + sr * 64 + sc,     kpp + (nbase + sr) * HID + sc);
        async_b128(kbuf[p] + sr * 64 + sc + 8, kpp + (nbase + sr) * HID + sc + 8);
        async_b128(vbuf[p] + sr * CW + sc,     vtp + sr * SEQ + nbase + sc);
        async_b128(vbuf[p] + sr * CW + sc + 8, vtp + sr * SEQ + nbase + sc + 8);
    }

    for (int nb = 0; nb < NCH; ++nb) {
        int bi = nb & 3;

        // Issue chunk nb+2 into ring slot (nb+2)&3 (safe: != nb-1, nb, nb+1 mod 4).
        if (nb + 2 < NCH) {
            int nn = (nb + 2) * CW, b2 = (nb + 2) & 3;
            async_b128(kbuf[b2] + sr * 64 + sc,     kpp + (nn + sr) * HID + sc);
            async_b128(kbuf[b2] + sr * 64 + sc + 8, kpp + (nn + sr) * HID + sc + 8);
            async_b128(vbuf[b2] + sr * CW + sc,     vtp + sr * SEQ + nn + sc);
            async_b128(vbuf[b2] + sr * CW + sc + 8, vtp + sr * SEQ + nn + sc + 8);
        }
        // Wait for chunk nb: 4 async ops per chunk per wave.
        if (nb + 2 < NCH)      asm volatile("s_wait_asynccnt 8" ::: "memory");
        else if (nb + 1 < NCH) asm volatile("s_wait_asynccnt 4" ::: "memory");
        else                   asm volatile("s_wait_asynccnt 0" ::: "memory");
        __syncthreads();

        // ---- S = (Q/8) K^T : four 16-col tiles from LDS K chunk ----
        v8f s0 = {}, s1 = {}, s2 = {}, s3 = {};
        {
            const u16* kc0 = kbuf[bi];
            int r0 = ( 0 + l16) * 64 + lh * 16;
            int r1 = (16 + l16) * 64 + lh * 16;
            int r2 = (32 + l16) * 64 + lh * 16;
            int r3 = (48 + l16) * 64 + lh * 16;
            s0 = wmma_bf16(q0, ldfrag(kc0, r0,      r0 + 8),  s0);
            s0 = wmma_bf16(q1, ldfrag(kc0, r0 + 32, r0 + 40), s0);
            s1 = wmma_bf16(q0, ldfrag(kc0, r1,      r1 + 8),  s1);
            s1 = wmma_bf16(q1, ldfrag(kc0, r1 + 32, r1 + 40), s1);
            s2 = wmma_bf16(q0, ldfrag(kc0, r2,      r2 + 8),  s2);
            s2 = wmma_bf16(q1, ldfrag(kc0, r2 + 32, r2 + 40), s2);
            s3 = wmma_bf16(q0, ldfrag(kc0, r3,      r3 + 8),  s3);
            s3 = wmma_bf16(q1, ldfrag(kc0, r3 + 32, r3 + 40), s3);
        }

        // ---- online softmax (rows live per VGPR-index per lane-half) ----
        float cm[8];
        #pragma unroll
        for (int v = 0; v < 8; ++v)
            cm[v] = fmaxf(fmaxf(s0[v], s1[v]), fmaxf(s2[v], s3[v]));
        #pragma unroll
        for (int v = 0; v < 8; ++v) {
            cm[v] = fmaxf(cm[v], __shfl_xor(cm[v], 1, 16));
            cm[v] = fmaxf(cm[v], __shfl_xor(cm[v], 2, 16));
            cm[v] = fmaxf(cm[v], __shfl_xor(cm[v], 4, 16));
            cm[v] = fmaxf(cm[v], __shfl_xor(cm[v], 8, 16));
        }
        float fc[8], rs[8];
        #pragma unroll
        for (int v = 0; v < 8; ++v) {
            float nm = fmaxf(mrow[v], cm[v]);
            fc[v] = __expf(mrow[v] - nm);
            mrow[v] = nm;
            s0[v] = __expf(s0[v] - nm);
            s1[v] = __expf(s1[v] - nm);
            s2[v] = __expf(s2[v] - nm);
            s3[v] = __expf(s3[v] - nm);
            rs[v] = (s0[v] + s1[v]) + (s2[v] + s3[v]);
        }
        #pragma unroll
        for (int v = 0; v < 8; ++v) {
            rs[v] += __shfl_xor(rs[v], 1, 16);
            rs[v] += __shfl_xor(rs[v], 2, 16);
            rs[v] += __shfl_xor(rs[v], 4, 16);
            rs[v] += __shfl_xor(rs[v], 8, 16);
            lrow[v] = lrow[v] * fc[v] + rs[v];
        }
        #pragma unroll
        for (int v = 0; v < 8; ++v) {
            o0[v] *= fc[v]; o1[v] *= fc[v]; o2[v] *= fc[v]; o3[v] *= fc[v];
        }

        // ---- P: C layout -> A layout via per-wave LDS tile (16 x 64) ----
        #pragma unroll
        for (int v = 0; v < 8; ++v) {
            int r = (v + lh * 8) * CW;
            pw[r +  0 + l16] = f2bf(s0[v]);
            pw[r + 16 + l16] = f2bf(s1[v]);
            pw[r + 32 + l16] = f2bf(s2[v]);
            pw[r + 48 + l16] = f2bf(s3[v]);
        }
        asm volatile("s_wait_dscnt 0" ::: "memory");
        int pb0 = l16 * CW + lh * 16;       // cols 0..31 of P
        int pb1 = l16 * CW + 32 + lh * 16;  // cols 32..63 of P
        v16bf pf0 = ldfrag(pw, pb0, pb0 + 8);
        v16bf pf1 = ldfrag(pw, pb1, pb1 + 8);

        // ---- O += P * V (4 d-groups x 2 ksteps) from LDS V chunk ----
        {
            const u16* vcb = vbuf[bi];
            int b0 = ( 0 + l16) * CW + lh * 16;
            int b1 = (16 + l16) * CW + lh * 16;
            int b2 = (32 + l16) * CW + lh * 16;
            int b3 = (48 + l16) * CW + lh * 16;
            o0 = wmma_bf16(pf0, ldfrag(vcb, b0,      b0 + 8),  o0);
            o0 = wmma_bf16(pf1, ldfrag(vcb, b0 + 32, b0 + 40), o0);
            o1 = wmma_bf16(pf0, ldfrag(vcb, b1,      b1 + 8),  o1);
            o1 = wmma_bf16(pf1, ldfrag(vcb, b1 + 32, b1 + 40), o1);
            o2 = wmma_bf16(pf0, ldfrag(vcb, b2,      b2 + 8),  o2);
            o2 = wmma_bf16(pf1, ldfrag(vcb, b2 + 32, b2 + 40), o2);
            o3 = wmma_bf16(pf0, ldfrag(vcb, b3,      b3 + 8),  o3);
            o3 = wmma_bf16(pf1, ldfrag(vcb, b3 + 32, b3 + 40), o3);
        }
    }

    // ---- normalize & store: Out[b, m, h*64 + d] ----
    #pragma unroll
    for (int v = 0; v < 8; ++v) {
        float inv = 1.0f / lrow[v];
        int m = mbase + v + lh * 8;
        size_t base = (size_t)(b * SEQ + m) * HID + h * HDIM + l16;
        Out[base + 0]  = f2bf(o0[v] * inv);
        Out[base + 16] = f2bf(o1[v] * inv);
        Out[base + 32] = f2bf(o2[v] * inv);
        Out[base + 48] = f2bf(o3[v] * inv);
    }
}

// ---------------- launch ----------------

extern "C" void kernel_launch(void* const* d_in, const int* in_sizes, int n_in,
                              void* d_out, int out_size, void* d_ws, size_t ws_size,
                              hipStream_t stream) {
    const float* q  = (const float*)d_in[0];
    const float* k  = (const float*)d_in[1];
    const float* vv = (const float*)d_in[2];
    const float* Wq = (const float*)d_in[3];
    const float* Wk = (const float*)d_in[4];
    const float* Wv = (const float*)d_in[5];
    const float* Wo = (const float*)d_in[6];

    u16* wsp = (u16*)d_ws;
    u16* qb  = wsp;
    u16* kb  = qb  + NQ_ELEMS;
    u16* vb  = kb  + NQ_ELEMS;
    u16* wqb = vb  + NQ_ELEMS;
    u16* wkb = wqb + NW_ELEMS;
    u16* wvb = wkb + NW_ELEMS;
    u16* wob = wvb + NW_ELEMS;
    u16* Qp  = wob + NW_ELEMS;
    u16* Kp  = Qp  + NQ_ELEMS;
    u16* Vt  = Kp  + NQ_ELEMS;
    u16* AO  = Vt  + NQ_ELEMS;

    const int nq = (int)NQ_ELEMS, nw = (int)NW_ELEMS;
    cvt_bf16_kernel<<<(nq + 255) / 256, 256, 0, stream>>>(q,  qb,  nq);
    cvt_bf16_kernel<<<(nq + 255) / 256, 256, 0, stream>>>(k,  kb,  nq);
    cvt_bf16_kernel<<<(nq + 255) / 256, 256, 0, stream>>>(vv, vb,  nq);
    cvt_bf16_kernel<<<(nw + 255) / 256, 256, 0, stream>>>(Wq, wqb, nw);
    cvt_bf16_kernel<<<(nw + 255) / 256, 256, 0, stream>>>(Wk, wkb, nw);
    cvt_bf16_kernel<<<(nw + 255) / 256, 256, 0, stream>>>(Wv, wvb, nw);
    cvt_bf16_kernel<<<(nw + 255) / 256, 256, 0, stream>>>(Wo, wob, nw);

    dim3 ggrid(HID / 128, MROWS / 128);   // (8, 32)
    gemm_bf16_kernel<3><<<ggrid, 256, 0, stream>>>(qb, wqb, Qp);   // Q pre-scaled
    gemm_bf16_kernel<0><<<ggrid, 256, 0, stream>>>(kb, wkb, Kp);
    gemm_bf16_kernel<1><<<ggrid, 256, 0, stream>>>(vb, wvb, Vt);

    attn_kernel<<<dim3(SEQ / 128, HEADS, BATCH), 256, 0, stream>>>(Qp, Kp, Vt, AO);

    gemm_bf16_kernel<2><<<ggrid, 256, 0, stream>>>(AO, wob, d_out);
}
